// Attention_27882927685970
// MI455X (gfx1250) — matrix operand order repaired and checked by reading, not compiled, not run on hardware
//
#include <hip/hip_runtime.h>

// ---------------------------------------------------------------------------
// CDNA5 (gfx1250) wave32 WMMA implementation of the mixed cos/cov attention.
// All GEMMs use v_wmma_f32_16x16x32_bf16 (bf16 inputs, fp32 accumulate).
// ---------------------------------------------------------------------------

#define DEV __device__ __forceinline__

typedef __bf16 bf16;
typedef __attribute__((ext_vector_type(8)))  __bf16 v8bf;
typedef __attribute__((ext_vector_type(16))) __bf16 v16bf;
typedef __attribute__((ext_vector_type(8)))  float  v8f;

#define C_QG   8
#define C_N    1024
#define C_DIM  512
#define C_H    8
#define C_D    64
#define C_TOK  (C_QG * C_N)   // 8192 tokens

// ---------------------------------------------------------------------------
// WMMA helpers (ISA 7.12.2 layouts, wave32)
// ---------------------------------------------------------------------------

DEV v8f wmma_bf16(v16bf a, v16bf b, v8f c) {
  // D = A(16x32 bf16) * B(32x16 bf16) + C(16x16 f32)
  return __builtin_amdgcn_wmma_f32_16x16x32_bf16(
      false, a, false, b, (short)0, c, false, false);
}

// A fragment: 16x32 bf16 tile, row-major with leading dim `ld`.
// lane (0..15): row m=lane, K 0..7 & 16..23 ; lane (16..31): row m=lane-16, K 8..15 & 24..31
DEV v16bf frag_a(const bf16* base, int ld, int lane) {
  const int m  = lane & 15;
  const int kh = (lane >> 4) & 1;
  const bf16* p = base + m * ld + kh * 8;
  v8bf lo = *(const v8bf*)p;          // K = kh*8 + 0..7
  v8bf hi = *(const v8bf*)(p + 16);   // K = 16 + kh*8 + 0..7
  return __builtin_shufflevector(lo, hi, 0,1,2,3,4,5,6,7,8,9,10,11,12,13,14,15);
}

// B fragment for B stored TRANSPOSED row-major: Brm[n][k] with B[k][n] = Brm[n][k].
// lane (0..15): col n=lane, K 0..7 & 16..23 ; lane (16..31): col n=lane-16, K 8..15 & 24..31
DEV v16bf frag_bt(const bf16* base, int ld, int lane) {
  const int n  = lane & 15;
  const int ks = ((lane >> 4) & 1) << 3;
  const bf16* p = base + n * ld + ks;
  v8bf lo = *(const v8bf*)p;
  v8bf hi = *(const v8bf*)(p + 16);
  return __builtin_shufflevector(lo, hi, 0,1,2,3,4,5,6,7,8,9,10,11,12,13,14,15);
}

DEV v8f zero8() { v8f z = {0.f,0.f,0.f,0.f,0.f,0.f,0.f,0.f}; return z; }

// ---------------------------------------------------------------------------
// Kernel 0: bf16-transpose the two 512x512 weight matrices.
// w_T[out][in] so B fragments are contiguous 32B reads.
// ---------------------------------------------------------------------------
__global__ void __launch_bounds__(256) k_prep(const float* __restrict__ w_in,
                                              const float* __restrict__ w_out,
                                              bf16* __restrict__ w_inT,
                                              bf16* __restrict__ w_outT) {
  const int idx = blockIdx.x * 256 + threadIdx.x;
  if (idx < C_DIM * C_DIM) {
    const int o = idx >> 9, i = idx & 511;
    w_inT [idx] = (bf16)w_in [(size_t)i * C_DIM + o];
    w_outT[idx] = (bf16)w_out[(size_t)i * C_DIM + o];
  }
}

// ---------------------------------------------------------------------------
// Kernel 1: LayerNorm + input projection (per mode: 0=q, 1=k, 2=v).
// Block: 256 thr (8 waves), 32 tokens x 512 outputs per block.
// Epilogue: f_q/f_k token-major bf16, f_v transposed ([h,qg,d,n]),
// per-token norms & feature sums, per-(h,d) sum/sumsq partials (deterministic).
// ---------------------------------------------------------------------------
__global__ void __launch_bounds__(256) k_proj(
    const float* __restrict__ xq, const float* __restrict__ xk,
    const float* __restrict__ xv,
    const float* __restrict__ ln_g, const float* __restrict__ ln_b,
    const bf16* __restrict__ w_inT,
    bf16* __restrict__ fq, bf16* __restrict__ fk, bf16* __restrict__ fvT,
    float* __restrict__ nq, float* __restrict__ sq,
    float* __restrict__ nk, float* __restrict__ sk,
    float* __restrict__ qps, float* __restrict__ qpq,
    float* __restrict__ kps, float* __restrict__ kpq) {
  __shared__ bf16  lnbuf[32 * C_DIM];   // normalized A tile (bf16), 32 KB
  __shared__ float ls_ths[32 * 8];      // per (row, head) feature sum
  __shared__ float ls_thss[32 * 8];     // per (row, head) sum of squares
  __shared__ float ls_hds[2 * 512];     // per (rt, col) sums
  __shared__ float ls_hdq[2 * 512];     // per (rt, col) sum of squares

  const int tid  = threadIdx.x;
  const int lane = tid & 31;
  const int wave = tid >> 5;
  const int kh   = (lane >> 4) & 1;
  const int mode = blockIdx.y;
  const int row0 = blockIdx.x * 32;
  const int qg   = row0 >> 10;
  const int n0   = row0 & 1023;

  const float* xin = (mode == 0) ? xq : ((mode == 1) ? xk : xv);

  // ---- LayerNorm: each wave handles 4 rows, lane-parallel over 512 cols ----
  for (int rr = 0; rr < 4; ++rr) {
    const int row = wave * 4 + rr;
    const float* xr = xin + (size_t)(row0 + row) * C_DIM;
    float4 vv[4];
    float s = 0.f, ss = 0.f;
#pragma unroll
    for (int j = 0; j < 4; ++j) {
      vv[j] = *(const float4*)(xr + lane * 16 + j * 4);
      s  += vv[j].x + vv[j].y + vv[j].z + vv[j].w;
      ss += vv[j].x * vv[j].x + vv[j].y * vv[j].y + vv[j].z * vv[j].z + vv[j].w * vv[j].w;
    }
#pragma unroll
    for (int m = 1; m < 32; m <<= 1) {
      s += __shfl_xor(s, m, 32);
      ss += __shfl_xor(ss, m, 32);
    }
    const float mu   = s * (1.f / 512.f);
    const float var  = ss * (1.f / 512.f) - mu * mu;
    const float rstd = rsqrtf(var + 1e-5f);
#pragma unroll
    for (int j = 0; j < 4; ++j) {
      const int c = lane * 16 + j * 4;
      const float4 g = *(const float4*)(ln_g + c);
      const float4 b = *(const float4*)(ln_b + c);
      lnbuf[row * C_DIM + c + 0] = (bf16)((vv[j].x - mu) * rstd * g.x + b.x);
      lnbuf[row * C_DIM + c + 1] = (bf16)((vv[j].y - mu) * rstd * g.y + b.y);
      lnbuf[row * C_DIM + c + 2] = (bf16)((vv[j].z - mu) * rstd * g.z + b.z);
      lnbuf[row * C_DIM + c + 3] = (bf16)((vv[j].w - mu) * rstd * g.w + b.w);
    }
  }
  __syncthreads();

  // ---- GEMM: C[32x512] = lnbuf @ w_inT^T  (WMMA bf16) ----
  // 16 (rt,h) groups; wave w handles groups w and w+8; each group = 4 col tiles.
  for (int g = wave; g < 16; g += 8) {
    const int rt = g >> 3;   // row tile (0..1)
    const int h  = g & 7;    // head
    float rs[8], rq[8];
    if (mode < 2) {
#pragma unroll
      for (int r = 0; r < 8; ++r) { rs[r] = 0.f; rq[r] = 0.f; }
    }
    for (int i = 0; i < 4; ++i) {
      const int ct = h * 4 + i;   // col tile (covers head h)
      v8f acc = zero8();
#pragma unroll
      for (int ks = 0; ks < 16; ++ks) {
        v16bf a = frag_a(lnbuf + rt * 16 * C_DIM + ks * 32, C_DIM, lane);
        v16bf b = frag_bt(w_inT + (size_t)ct * 16 * C_DIM + ks * 32, C_DIM, lane);
        acc = wmma_bf16(a, b, acc);
      }
      const int c = ct * 16 + (lane & 15);
      const int d = c & 63;
      float cs = 0.f, css = 0.f;
#pragma unroll
      for (int r = 0; r < 8; ++r) {
        const float f = acc[r];
        const int rowL = rt * 16 + r + kh * 8;
        const int n = n0 + rowL;
        if (mode < 2) {
          cs += f; css += f * f;
          float a1 = f, a2 = f * f;
#pragma unroll
          for (int m = 1; m < 16; m <<= 1) {
            a1 += __shfl_xor(a1, m, 32);
            a2 += __shfl_xor(a2, m, 32);
          }
          rs[r] += a1; rq[r] += a2;
        }
        const size_t hb = (size_t)(h * C_QG + qg);
        if (mode == 0)      fq [(hb * C_N + n) * C_D + d] = (bf16)f;
        else if (mode == 1) fk [(hb * C_N + n) * C_D + d] = (bf16)f;
        else                fvT[(hb * C_D + d) * C_N + n] = (bf16)f;
      }
      if (mode < 2) {
        cs  += __shfl_xor(cs, 16, 32);
        css += __shfl_xor(css, 16, 32);
        if (lane < 16) { ls_hds[rt * 512 + c] = cs; ls_hdq[rt * 512 + c] = css; }
      }
    }
    if (mode < 2 && (lane & 15) == 0) {
#pragma unroll
      for (int r = 0; r < 8; ++r) {
        const int rowL = rt * 16 + r + kh * 8;
        ls_ths[rowL * 8 + h]  = rs[r];
        ls_thss[rowL * 8 + h] = rq[r];
      }
    }
  }
  __syncthreads();

  // ---- flush per-token norms/sums and per-(h,d) block partials ----
  if (mode < 2) {
    float* nrm = (mode == 0) ? nq : nk;
    float* fsm = (mode == 0) ? sq : sk;
    {
      const int rowL = tid >> 3, h = tid & 7;
      const int n = n0 + rowL;
      const size_t o = (size_t)(h * C_QG + qg) * C_N + n;
      fsm[o] = ls_ths[tid];
      nrm[o] = sqrtf(ls_thss[tid]);
    }
    float* ps = (mode == 0) ? qps : kps;
    float* pq = (mode == 0) ? qpq : kpq;
    for (int i = tid; i < 512; i += 256) {
      ps[(size_t)blockIdx.x * 512 + i] = ls_hds[i] + ls_hds[512 + i];
      pq[(size_t)blockIdx.x * 512 + i] = ls_hdq[i] + ls_hdq[512 + i];
    }
  }
}

// ---------------------------------------------------------------------------
// Kernel 2: variance penalty + per-head gate MLP (tiny; single block).
// ---------------------------------------------------------------------------
__global__ void __launch_bounds__(64) k_gate(
    const float* __restrict__ qps, const float* __restrict__ qpq,
    const float* __restrict__ kps, const float* __restrict__ kpq,
    const float* __restrict__ p_w1, const float* __restrict__ p_b1,
    const float* __restrict__ p_ln_g, const float* __restrict__ p_ln_b,
    const float* __restrict__ p_w2, const float* __restrict__ p_b2,
    float* __restrict__ wgate) {
  __shared__ float qs_hd[512], qq_hd[512], ks_hd[512], kq_hd[512];
  __shared__ float red[64], yv[64], xv[128], sh[4];
  const int t = threadIdx.x;  // 0..63 (= d)

  // reduce 256 block partials -> per-(h,d) totals
  for (int h = 0; h < 8; ++h) {
    const int c = h * 64 + t;
    float s1 = 0.f, s2 = 0.f, s3 = 0.f, s4 = 0.f;
    for (int b = 0; b < 256; ++b) {
      s1 += qps[(size_t)b * 512 + c];
      s2 += qpq[(size_t)b * 512 + c];
      s3 += kps[(size_t)b * 512 + c];
      s4 += kpq[(size_t)b * 512 + c];
    }
    qs_hd[c] = s1; qq_hd[c] = s2; ks_hd[c] = s3; kq_hd[c] = s4;
  }
  __syncthreads();

  // per-d variance across ALL heads*tokens (M = H*QG*N rows), ddof=1
  float sumq = 0.f, sqq = 0.f, sumk = 0.f, sqk = 0.f;
  for (int h = 0; h < 8; ++h) {
    sumq += qs_hd[h * 64 + t]; sqq += qq_hd[h * 64 + t];
    sumk += ks_hd[h * 64 + t]; sqk += kq_hd[h * 64 + t];
  }
  const float M = (float)(C_H * C_QG * C_N);
  const float varq = (sqq - sumq * sumq / M) / (M - 1.f);
  const float vark = (sqk - sumk * sumk / M) / (M - 1.f);
  red[t] = fmaxf(1.f - sqrtf(varq + 1e-4f), 0.f) +
           fmaxf(1.f - sqrtf(vark + 1e-4f), 0.f);
  __syncthreads();
  if (t == 0) {
    float s = 0.f;
    for (int i = 0; i < 64; ++i) s += red[i];
    sh[2] = s * (1.f / 64.f);
  }
  __syncthreads();
  const float vpen = sh[2];

  // per-head gate MLP: sigmoid(relu(LN(x @ p_w1 + b1)) @ p_w2 + b2)
  for (int h = 0; h < 8; ++h) {
    xv[t]      = qs_hd[h * 64 + t] * (1.f / (float)(C_QG * C_N));
    xv[64 + t] = ks_hd[h * 64 + t] * (1.f / (float)(C_QG * C_N));
    __syncthreads();
    float y = p_b1[t];
    for (int i = 0; i < 128; ++i) y += xv[i] * p_w1[i * 64 + t];
    yv[t] = y;
    __syncthreads();
    if (t == 0) {
      float mu = 0.f;
      for (int i = 0; i < 64; ++i) mu += yv[i];
      mu *= (1.f / 64.f);
      float va = 0.f;
      for (int i = 0; i < 64; ++i) { const float dd = yv[i] - mu; va += dd * dd; }
      va *= (1.f / 64.f);
      sh[0] = mu; sh[1] = rsqrtf(va + 1e-5f);
    }
    __syncthreads();
    const float h1 = fmaxf((y - sh[0]) * sh[1] * p_ln_g[t] + p_ln_b[t], 0.f);
    red[t] = h1 * p_w2[t];
    __syncthreads();
    if (t == 0) {
      float s = p_b2[0];
      for (int i = 0; i < 64; ++i) s += red[i];
      const float wg = 1.f / (1.f + __expf(-s));
      wgate[h] = wg / (1.f + vpen);
    }
    __syncthreads();
  }
}

// ---------------------------------------------------------------------------
// Kernel 3: attention core. One block per (h*qg, 64-row tile).
// S = Q·K^T (WMMA); mixed = (1-w)*S/(|q||k|) + w*(S - sq*sk/D)/D ;
// O += mixed · V (WMMA) with V staged transposed.
// ---------------------------------------------------------------------------
__global__ void __launch_bounds__(256) k_attn(
    const bf16* __restrict__ fq, const bf16* __restrict__ fk,
    const bf16* __restrict__ fvT,
    const float* __restrict__ nq, const float* __restrict__ sq,
    const float* __restrict__ nk, const float* __restrict__ sk,
    const float* __restrict__ wgate, bf16* __restrict__ oatt) {
  __shared__ bf16 Qs[64 * 64];
  __shared__ bf16 Ks[64 * 64];
  __shared__ bf16 Vs[64 * 64];   // V^T tile: [d][m]
  __shared__ bf16 Ss[64 * 64];   // mixed score tile (bf16)
  __shared__ float nqv[64], sqv[64], nkv[64], skv[64];

  const int tid  = threadIdx.x;
  const int lane = tid & 31;
  const int wave = tid >> 5;
  const int kh   = (lane >> 4) & 1;
  const int hq   = blockIdx.y;         // h*QG + qg
  const int h    = hq >> 3;
  const int qg   = hq & 7;
  const int row0 = blockIdx.x * 64;
  const size_t bo = (size_t)hq * C_N * C_D;  // == hq * D * N as well

  const float w  = wgate[h];
  const float cw = 1.f - w;

  // stage Q tile + its per-row stats
#pragma unroll
  for (int it = 0; it < 2; ++it) {
    const int chunk = tid + it * 256;         // 0..511
    const int r = chunk >> 3, c8 = (chunk & 7) * 8;
    *(v8bf*)&Qs[r * 64 + c8] = *(const v8bf*)&fq[bo + (size_t)(row0 + r) * C_D + c8];
  }
  if (tid < 64) {
    nqv[tid] = nq[(size_t)hq * C_N + row0 + tid];
    sqv[tid] = sq[(size_t)hq * C_N + row0 + tid];
  }

  const int t0 = wave * 2;   // each wave owns 2 of the 16 output tiles
  v8f oacc[2];
  oacc[0] = zero8();
  oacc[1] = zero8();

  for (int ms = 0; ms < 16; ++ms) {
    const int mb = ms * 64;
    __syncthreads();
#pragma unroll
    for (int it = 0; it < 2; ++it) {
      const int chunk = tid + it * 256;
      const int r = chunk >> 3, c8 = (chunk & 7) * 8;
      *(v8bf*)&Ks[r * 64 + c8] = *(const v8bf*)&fk[bo + (size_t)(mb + r) * C_D + c8];
      *(v8bf*)&Vs[r * 64 + c8] = *(const v8bf*)&fvT[bo + (size_t)r * C_N + mb + c8];
    }
    if (tid < 64) {
      nkv[tid] = nk[(size_t)hq * C_N + mb + tid];
      skv[tid] = sk[(size_t)hq * C_N + mb + tid];
    }
    __syncthreads();

    // S = Q · K^T, then cos/cov mix -> Ss (bf16)
#pragma unroll
    for (int t = 0; t < 2; ++t) {
      const int tile = t0 + t;
      const int rt = tile >> 2, ct = tile & 3;
      v8f s = zero8();
#pragma unroll
      for (int ks = 0; ks < 2; ++ks) {
        v16bf a = frag_a(Qs + rt * 16 * 64 + ks * 32, 64, lane);
        v16bf b = frag_bt(Ks + ct * 16 * 64 + ks * 32, 64, lane);
        s = wmma_bf16(a, b, s);
      }
      const int colL = ct * 16 + (lane & 15);
      const float rnk = 1.f / nkv[colL];
      const float skc = skv[colL];
#pragma unroll
      for (int r = 0; r < 8; ++r) {
        const int rowL = rt * 16 + r + kh * 8;
        const float f = s[r];
        const float cosv = f * (1.f / nqv[rowL]) * rnk;
        const float covv = (f - sqv[rowL] * skc * (1.f / 64.f)) * (1.f / 64.f);
        Ss[rowL * 64 + colL] = (bf16)(cw * cosv + w * covv);
      }
    }
    __syncthreads();

    // O += mixed · V  (B from V^T tile -> contiguous reads)
#pragma unroll
    for (int t = 0; t < 2; ++t) {
      const int tile = t0 + t;
      const int rt = tile >> 2, ct = tile & 3;
#pragma unroll
      for (int ks = 0; ks < 2; ++ks) {
        v16bf a = frag_a(Ss + rt * 16 * 64 + ks * 32, 64, lane);
        v16bf b = frag_bt(Vs + ct * 16 * 64 + ks * 32, 64, lane);
        oacc[t] = wmma_bf16(a, b, oacc[t]);
      }
    }
  }

  // store O tile to out_attn[qg][n][h*64 + d] (bf16)
#pragma unroll
  for (int t = 0; t < 2; ++t) {
    const int tile = t0 + t;
    const int rt = tile >> 2, ct = tile & 3;
    const int dcol = ct * 16 + (lane & 15);
#pragma unroll
    for (int r = 0; r < 8; ++r) {
      const int rowL = rt * 16 + r + kh * 8;
      const size_t n = (size_t)(row0 + rowL);
      oatt[((size_t)qg * C_N + n) * (C_H * C_D) + h * C_D + dcol] = (bf16)oacc[t][r];
    }
  }
}

// ---------------------------------------------------------------------------
// Kernel 4: output projection out = oatt @ w_out + b_out (fp32 result).
// ---------------------------------------------------------------------------
__global__ void __launch_bounds__(256) k_outproj(
    const bf16* __restrict__ oatt, const bf16* __restrict__ w_outT,
    const float* __restrict__ b_out, float* __restrict__ out) {
  const int tid  = threadIdx.x;
  const int lane = tid & 31;
  const int wave = tid >> 5;
  const int kh   = (lane >> 4) & 1;
  const int row0 = blockIdx.x * 32;

  for (int tile = wave; tile < 64; tile += 8) {
    const int rt = tile >> 5, ct = tile & 31;
    v8f acc = zero8();
#pragma unroll
    for (int ks = 0; ks < 16; ++ks) {
      v16bf a = frag_a(oatt + (size_t)(row0 + rt * 16) * C_DIM + ks * 32, C_DIM, lane);
      v16bf b = frag_bt(w_outT + (size_t)ct * 16 * C_DIM + ks * 32, C_DIM, lane);
      acc = wmma_bf16(a, b, acc);
    }
    const int c = ct * 16 + (lane & 15);
    const float bias = b_out[c];
#pragma unroll
    for (int r = 0; r < 8; ++r) {
      const int rowL = rt * 16 + r + kh * 8;
      out[(size_t)(row0 + rowL) * C_DIM + c] = acc[r] + bias;
    }
  }
}

// ---------------------------------------------------------------------------
// Host launcher
// ---------------------------------------------------------------------------
extern "C" void kernel_launch(void* const* d_in, const int* in_sizes, int n_in,
                              void* d_out, int out_size, void* d_ws, size_t ws_size,
                              hipStream_t stream) {
  (void)in_sizes; (void)n_in; (void)out_size; (void)ws_size;
  const float* q      = (const float*)d_in[0];
  const float* k      = (const float*)d_in[1];
  const float* v      = (const float*)d_in[2];
  const float* ln_g   = (const float*)d_in[3];
  const float* ln_b   = (const float*)d_in[4];
  const float* w_in   = (const float*)d_in[5];
  const float* p_w1   = (const float*)d_in[6];
  const float* p_b1   = (const float*)d_in[7];
  const float* p_ln_g = (const float*)d_in[8];
  const float* p_ln_b = (const float*)d_in[9];
  const float* p_w2   = (const float*)d_in[10];
  const float* p_b2   = (const float*)d_in[11];
  const float* w_out  = (const float*)d_in[12];
  const float* b_out  = (const float*)d_in[13];
  float* out = (float*)d_out;

  char* ws = (char*)d_ws;
  size_t off = 0;
  auto carve = [&](size_t bytes) -> void* {
    void* p = ws + off;
    off += (bytes + 255) & ~(size_t)255;
    return p;
  };
  const size_t feat_elems = (size_t)C_H * C_QG * C_N * C_D;   // 4M
  bf16*  fq     = (bf16*)carve(sizeof(bf16) * feat_elems);
  bf16*  fk     = (bf16*)carve(sizeof(bf16) * feat_elems);
  bf16*  fvT    = (bf16*)carve(sizeof(bf16) * feat_elems);
  bf16*  oatt   = (bf16*)carve(sizeof(bf16) * (size_t)C_QG * C_N * C_H * C_D);
  bf16*  w_inT  = (bf16*)carve(sizeof(bf16) * C_DIM * C_DIM);
  bf16*  w_outT = (bf16*)carve(sizeof(bf16) * C_DIM * C_DIM);
  float* nq     = (float*)carve(sizeof(float) * (size_t)C_H * C_QG * C_N);
  float* sq     = (float*)carve(sizeof(float) * (size_t)C_H * C_QG * C_N);
  float* nk     = (float*)carve(sizeof(float) * (size_t)C_H * C_QG * C_N);
  float* sk     = (float*)carve(sizeof(float) * (size_t)C_H * C_QG * C_N);
  float* qps    = (float*)carve(sizeof(float) * 256 * 512);
  float* qpq    = (float*)carve(sizeof(float) * 256 * 512);
  float* kps    = (float*)carve(sizeof(float) * 256 * 512);
  float* kpq    = (float*)carve(sizeof(float) * 256 * 512);
  float* wg     = (float*)carve(sizeof(float) * 8);

  k_prep<<<dim3((C_DIM * C_DIM + 255) / 256), dim3(256), 0, stream>>>(
      w_in, w_out, w_inT, w_outT);

  k_proj<<<dim3(C_TOK / 32, 3), dim3(256), 0, stream>>>(
      q, k, v, ln_g, ln_b, w_inT, fq, fk, fvT,
      nq, sq, nk, sk, qps, qpq, kps, kpq);

  k_gate<<<dim3(1), dim3(64), 0, stream>>>(
      qps, qpq, kps, kpq, p_w1, p_b1, p_ln_g, p_ln_b, p_w2, p_b2, wg);

  k_attn<<<dim3(C_N / 64, C_H * C_QG), dim3(256), 0, stream>>>(
      fq, fk, fvT, nq, sq, nk, sk, wg, oatt);

  k_outproj<<<dim3(C_TOK / 32), dim3(256), 0, stream>>>(
      oatt, w_outT, b_out, out);
}
